// ContextualActorSpike_22144851378858
// MI455X (gfx1250) — 16-bit and 8-bit matrix operand order repaired and checked by reading, compiled, not run on hardware
//
#include <hip/hip_runtime.h>
#include <hip/hip_bf16.h>
#include <math.h>

typedef __attribute__((ext_vector_type(16))) _Float16 v16h;
typedef __attribute__((ext_vector_type(8)))  _Float16 v8h;
typedef __attribute__((ext_vector_type(8)))  float    v8f;
typedef __attribute__((ext_vector_type(16))) int      v16i;

#define HID      1024
#define IN_DIM   192
#define OBS_DIM  128
#define CTX_DIM  64
#define ACT_DIM  32
#define T_STEPS  4
#define W8_SCALE     64.0f
#define INV_W8_SCALE (1.0f / 64.0f)
#define FP8_ONE  0x38

__device__ __forceinline__ void async_cp16(const void* gsrc, void* lds) {
    unsigned loff = (unsigned)(size_t)lds;
    unsigned long long g = (unsigned long long)(size_t)gsrc;
    asm volatile("global_load_async_to_lds_b128 %0, %1, off"
                 :: "v"(loff), "v"(g) : "memory");
}
__device__ __forceinline__ void async_cp8(const void* gsrc, void* lds) {
    unsigned loff = (unsigned)(size_t)lds;
    unsigned long long g = (unsigned long long)(size_t)gsrc;
    asm volatile("global_load_async_to_lds_b64 %0, %1, off"
                 :: "v"(loff), "v"(g) : "memory");
}
#define WAIT_ASYNC(n) asm volatile("s_wait_asynccnt %0" :: "n"(n) : "memory")

__device__ __forceinline__ unsigned char f32_to_e4m3(float x) {
    unsigned int u = __float_as_uint(x);
    unsigned char sign = (unsigned char)((u >> 24) & 0x80u);
    float ax = fabsf(x);
    if (!(ax == ax)) return (unsigned char)(sign | 0x7F);
    if (ax >= 448.0f) return (unsigned char)(sign | 0x7E);
    if (ax < 0x1p-10f) return sign;
    int ei = (int)((u >> 23) & 0xFF) - 127;
    int expn = ei < -6 ? -6 : ei;
    float ulp = exp2f((float)(expn - 3));
    float q = rintf(ax / ulp) * ulp;
    if (q >= 448.0f) return (unsigned char)(sign | 0x7E);
    unsigned int qu = __float_as_uint(q);
    int qe = (int)((qu >> 23) & 0xFF) - 127;
    unsigned char bits;
    if (q < 0x1p-6f) {
        bits = (unsigned char)rintf(q * 512.0f);
    } else {
        unsigned int mant = (qu >> 20) & 0x7u;
        bits = (unsigned char)(((qe + 7) << 3) | mant);
    }
    return (unsigned char)(sign | bits);
}

__global__ void k_pack_x(const float* __restrict__ obs, const float* __restrict__ ctx,
                         _Float16* __restrict__ x16, int B) {
    int i = blockIdx.x * 256 + threadIdx.x;
    if (i >= B * IN_DIM) return;
    int b = i / IN_DIM, c = i % IN_DIM;
    float v = (c < OBS_DIM) ? obs[(size_t)b * OBS_DIM + c]
                            : ctx[(size_t)b * CTX_DIM + (c - OBS_DIM)];
    x16[i] = (_Float16)v;
}

__global__ void k_cvt_f16(const float* __restrict__ src, _Float16* __restrict__ dst, int n) {
    int i = blockIdx.x * 256 + threadIdx.x;
    if (i < n) dst[i] = (_Float16)src[i];
}

__global__ void k_cvt_fp8(const float* __restrict__ src, unsigned char* __restrict__ dst,
                          int n, float scale) {
    int i = blockIdx.x * 256 + threadIdx.x;
    if (i < n) dst[i] = f32_to_e4m3(src[i] * scale);
}

__global__ __launch_bounds__(256) void k_layer1(const _Float16* __restrict__ x16,
                                                const _Float16* __restrict__ W1h,
                                                const float* __restrict__ b1,
                                                unsigned char* __restrict__ s1, int B) {
    __shared__ __align__(16) _Float16 As[2][128 * 32];
    __shared__ __align__(16) _Float16 Bs[2][64 * 32];

    const int tid   = threadIdx.x;
    const int wave  = tid >> 5;
    const int lane  = tid & 31;
    const int hlf   = lane >> 4;
    const int r     = lane & 15;
    const int rowB  = blockIdx.x * 128;
    const int nB    = blockIdx.y * 64;
    const int mB    = wave * 16;

    auto issue = [&](int k0, int buf) {
        #pragma unroll
        for (int j = 0; j < 2; ++j) {
            int c = tid + j * 256;
            int row = c >> 2, col = c & 3;
            async_cp16(x16 + (size_t)(rowB + row) * IN_DIM + k0 + col * 8,
                       &As[buf][row * 32 + col * 8]);
        }
        {
            int n = tid >> 2, col = tid & 3;
            async_cp16(W1h + (size_t)(nB + n) * IN_DIM + k0 + col * 8,
                       &Bs[buf][n * 32 + col * 8]);
        }
    };

    v8f acc[4] = {};
    const int NK = IN_DIM / 32;
    issue(0, 0);
    for (int kc = 0; kc < NK; ++kc) {
        if (kc + 1 < NK) { issue((kc + 1) * 32, (kc + 1) & 1); WAIT_ASYNC(3); }
        else             { WAIT_ASYNC(0); }
        __syncthreads();

        const int buf = kc & 1;
        v16h a;
        {
            const _Float16* rp = &As[buf][(mB + r) * 32];
            v8h lo = *(const v8h*)(rp + hlf * 8);
            v8h hi = *(const v8h*)(rp + 16 + hlf * 8);
            #pragma unroll
            for (int i = 0; i < 8; ++i) { a[i] = lo[i]; a[8 + i] = hi[i]; }
        }
        #pragma unroll
        for (int ns = 0; ns < 4; ++ns) {
            v16h bf;
            const _Float16* np = &Bs[buf][(ns * 16 + r) * 32];
            v8h lo = *(const v8h*)(np + hlf * 8);
            v8h hi = *(const v8h*)(np + 16 + hlf * 8);
            #pragma unroll
            for (int i = 0; i < 8; ++i) { bf[i] = lo[i]; bf[8 + i] = hi[i]; }
            acc[ns] = __builtin_amdgcn_wmma_f32_16x16x32_f16(false, a, false, bf,
                                                             (short)0, acc[ns], false, false);
        }
        __syncthreads();
    }

    #pragma unroll
    for (int ns = 0; ns < 4; ++ns) {
        const int gn = nB + ns * 16 + r;
        const float bias = b1[gn];
        #pragma unroll
        for (int e = 0; e < 8; ++e) {
            const int gb = rowB + mB + e + 8 * hlf;
            float u = acc[ns][e] + bias;
            float v = 0.0f;
            #pragma unroll
            for (int t = 0; t < T_STEPS; ++t) {
                v = 0.5f * (v + u);
                float s = (v >= 1.0f) ? 1.0f : 0.0f;
                s1[((size_t)t * B + gb) * HID + gn] = (s != 0.0f) ? (unsigned char)FP8_ONE : 0;
                v *= (1.0f - s);
            }
        }
    }
}

template <bool WRITE_FEAT>
__global__ __launch_bounds__(256) void k_layer_spike(const unsigned char* __restrict__ sPrev,
                                                     const unsigned char* __restrict__ Wq,
                                                     const float* __restrict__ bias,
                                                     unsigned char* __restrict__ sOut,
                                                     _Float16* __restrict__ feat, int B) {
    __shared__ __align__(16) unsigned char As[2][128 * 128];
    __shared__ __align__(16) unsigned char Bs[2][64 * 128];

    const int tid  = threadIdx.x;
    const int wave = tid >> 5;
    const int lane = tid & 31;
    const int hlf  = lane >> 4;
    const int r    = lane & 15;
    const int rowB = blockIdx.x * 128;
    const int nB   = blockIdx.y * 64;
    const int mB   = wave * 16;

    auto issue = [&](int t, int k0, int buf) {
        #pragma unroll
        for (int j = 0; j < 4; ++j) {
            int c = tid + j * 256;
            int row = c >> 3, col = c & 7;
            async_cp16(sPrev + ((size_t)t * B + rowB + row) * HID + k0 + col * 16,
                       &As[buf][row * 128 + col * 16]);
        }
        #pragma unroll
        for (int j = 0; j < 2; ++j) {
            int c = tid + j * 256;
            int n = c >> 3, col = c & 7;
            async_cp16(Wq + (size_t)(nB + n) * HID + k0 + col * 16,
                       &Bs[buf][n * 128 + col * 16]);
        }
    };

    float vmem[4][8];
    float fsum[4][8];
    #pragma unroll
    for (int ns = 0; ns < 4; ++ns)
        #pragma unroll
        for (int e = 0; e < 8; ++e) { vmem[ns][e] = 0.0f; fsum[ns][e] = 0.0f; }

    const int NK = HID / 128;
    for (int t = 0; t < T_STEPS; ++t) {
        v8f acc[4] = {};
        issue(t, 0, 0);
        for (int kc = 0; kc < NK; ++kc) {
            if (kc + 1 < NK) { issue(t, (kc + 1) * 128, (kc + 1) & 1); WAIT_ASYNC(6); }
            else             { WAIT_ASYNC(0); }
            __syncthreads();

            const int buf = kc & 1;
            v16i a;
            {
                const unsigned char* rp = &As[buf][(mB + r) * 128];
                #pragma unroll
                for (int j = 0; j < 4; ++j) {
                    int2 p0 = *(const int2*)(rp + 32 * (j >> 1) + 16 * hlf + 8 * (j & 1));
                    a[2 * j]     = p0.x; a[2 * j + 1] = p0.y;
                    int2 p1 = *(const int2*)(rp + 64 + 32 * (j >> 1) + 16 * hlf + 8 * (j & 1));
                    a[8 + 2 * j] = p1.x; a[9 + 2 * j] = p1.y;
                }
            }
            #pragma unroll
            for (int ns = 0; ns < 4; ++ns) {
                v16i bf;
                const unsigned char* np = &Bs[buf][(ns * 16 + r) * 128];
                #pragma unroll
                for (int c2 = 0; c2 < 4; ++c2) {
                    int4 q = *(const int4*)(np + 32 * c2 + 16 * hlf);
                    bf[4 * c2] = q.x; bf[4 * c2 + 1] = q.y;
                    bf[4 * c2 + 2] = q.z; bf[4 * c2 + 3] = q.w;
                }
                acc[ns] = __builtin_amdgcn_wmma_f32_16x16x128_fp8_fp8(a, bf, (short)0,
                                                                      acc[ns], false, false);
            }
            __syncthreads();
        }

        #pragma unroll
        for (int ns = 0; ns < 4; ++ns) {
            const int gn = nB + ns * 16 + r;
            const float bs = bias[gn];
            #pragma unroll
            for (int e = 0; e < 8; ++e) {
                const int gb = rowB + mB + e + 8 * hlf;
                float u = acc[ns][e] * INV_W8_SCALE + bs;
                float v = vmem[ns][e];
                v = 0.5f * (v + u);
                float s = (v >= 1.0f) ? 1.0f : 0.0f;
                if (!WRITE_FEAT)
                    sOut[((size_t)t * B + gb) * HID + gn] = (s != 0.0f) ? (unsigned char)FP8_ONE : 0;
                else
                    fsum[ns][e] += s;
                vmem[ns][e] = v * (1.0f - s);
            }
        }
    }

    if (WRITE_FEAT) {
        #pragma unroll
        for (int ns = 0; ns < 4; ++ns) {
            const int gn = nB + ns * 16 + r;
            #pragma unroll
            for (int e = 0; e < 8; ++e) {
                const int gb = rowB + mB + e + 8 * hlf;
                feat[(size_t)gb * HID + gn] = (_Float16)(fsum[ns][e] * 0.25f);
            }
        }
    }
}

__global__ __launch_bounds__(256) void k_head(const _Float16* __restrict__ feat,
                                              const _Float16* __restrict__ Wmh,
                                              const float* __restrict__ bm,
                                              const float* __restrict__ noise,
                                              float* __restrict__ out, int B) {
    __shared__ __align__(16) _Float16 As[2][128 * 32];
    __shared__ __align__(16) _Float16 Bs[2][32 * 32];

    const int tid  = threadIdx.x;
    const int wave = tid >> 5;
    const int lane = tid & 31;
    const int hlf  = lane >> 4;
    const int r    = lane & 15;
    const int rowB = blockIdx.x * 128;
    const int mB   = wave * 16;

    auto issue = [&](int k0, int buf) {
        #pragma unroll
        for (int j = 0; j < 2; ++j) {
            int c = tid + j * 256;
            int row = c >> 2, col = c & 3;
            async_cp16(feat + (size_t)(rowB + row) * HID + k0 + col * 8,
                       &As[buf][row * 32 + col * 8]);
        }
        {
            int n = tid >> 3, col = tid & 7;
            async_cp8(Wmh + (size_t)n * HID + k0 + col * 4,
                      &Bs[buf][n * 32 + col * 4]);
        }
    };

    v8f acc[2] = {};
    const int NK = HID / 32;
    issue(0, 0);
    for (int kc = 0; kc < NK; ++kc) {
        if (kc + 1 < NK) { issue((kc + 1) * 32, (kc + 1) & 1); WAIT_ASYNC(3); }
        else             { WAIT_ASYNC(0); }
        __syncthreads();

        const int buf = kc & 1;
        v16h a;
        {
            const _Float16* rp = &As[buf][(mB + r) * 32];
            v8h lo = *(const v8h*)(rp + hlf * 8);
            v8h hi = *(const v8h*)(rp + 16 + hlf * 8);
            #pragma unroll
            for (int i = 0; i < 8; ++i) { a[i] = lo[i]; a[8 + i] = hi[i]; }
        }
        #pragma unroll
        for (int ns = 0; ns < 2; ++ns) {
            v16h bf;
            const _Float16* np = &Bs[buf][(ns * 16 + r) * 32];
            v8h lo = *(const v8h*)(np + hlf * 8);
            v8h hi = *(const v8h*)(np + 16 + hlf * 8);
            #pragma unroll
            for (int i = 0; i < 8; ++i) { bf[i] = lo[i]; bf[8 + i] = hi[i]; }
            acc[ns] = __builtin_amdgcn_wmma_f32_16x16x32_f16(false, a, false, bf,
                                                             (short)0, acc[ns], false, false);
        }
        __syncthreads();
    }

    #pragma unroll
    for (int ns = 0; ns < 2; ++ns) {
        const int gn = ns * 16 + r;
        const float bs = bm[gn];
        const float nz = fminf(fmaxf(noise[gn], -0.1f), 0.1f);
        #pragma unroll
        for (int e = 0; e < 8; ++e) {
            const int gb = rowB + mB + e + 8 * hlf;
            float am = tanhf(acc[ns][e] + bs);
            out[(size_t)gb * ACT_DIM + gn] = am;
            out[(size_t)B * ACT_DIM + (size_t)gb * ACT_DIM + gn] = am + nz;
        }
    }
}

extern "C" void kernel_launch(void* const* d_in, const int* in_sizes, int n_in,
                              void* d_out, int out_size, void* d_ws, size_t ws_size,
                              hipStream_t stream) {
    const float* obs   = (const float*)d_in[0];
    const float* ctx   = (const float*)d_in[1];
    const float* noise = (const float*)d_in[2];
    const float* W1    = (const float*)d_in[3];
    const float* b1    = (const float*)d_in[4];
    const float* W2    = (const float*)d_in[5];
    const float* b2    = (const float*)d_in[6];
    const float* W3    = (const float*)d_in[7];
    const float* b3    = (const float*)d_in[8];
    const float* Wm    = (const float*)d_in[9];
    const float* bm    = (const float*)d_in[10];
    (void)n_in; (void)out_size; (void)ws_size;

    const int B = in_sizes[0] / OBS_DIM;

    char* base = (char*)d_ws;
    size_t o = 0;
    auto carve = [&](size_t bytes) -> void* {
        void* q = base + o;
        o += (bytes + 255) & ~(size_t)255;
        return q;
    };
    _Float16*      x16  = (_Float16*)carve((size_t)B * IN_DIM * sizeof(_Float16));
    _Float16*      W1h  = (_Float16*)carve((size_t)HID * IN_DIM * sizeof(_Float16));
    unsigned char* W2q  = (unsigned char*)carve((size_t)HID * HID);
    unsigned char* W3q  = (unsigned char*)carve((size_t)HID * HID);
    _Float16*      Wmh  = (_Float16*)carve((size_t)ACT_DIM * HID * sizeof(_Float16));
    unsigned char* s1   = (unsigned char*)carve((size_t)T_STEPS * B * HID);
    unsigned char* s2   = (unsigned char*)carve((size_t)T_STEPS * B * HID);
    _Float16*      feat = (_Float16*)carve((size_t)B * HID * sizeof(_Float16));

    k_pack_x<<<(B * IN_DIM + 255) / 256, 256, 0, stream>>>(obs, ctx, x16, B);
    k_cvt_f16<<<(HID * IN_DIM + 255) / 256, 256, 0, stream>>>(W1, W1h, HID * IN_DIM);
    k_cvt_f16<<<(ACT_DIM * HID + 255) / 256, 256, 0, stream>>>(Wm, Wmh, ACT_DIM * HID);
    k_cvt_fp8<<<(HID * HID + 255) / 256, 256, 0, stream>>>(W2, W2q, HID * HID, W8_SCALE);
    k_cvt_fp8<<<(HID * HID + 255) / 256, 256, 0, stream>>>(W3, W3q, HID * HID, W8_SCALE);

    dim3 gridL(B / 128, HID / 64);
    k_layer1<<<gridL, 256, 0, stream>>>(x16, W1h, b1, s1, B);
    k_layer_spike<false><<<gridL, 256, 0, stream>>>(s1, W2q, b2, s2, nullptr, B);
    k_layer_spike<true><<<gridL, 256, 0, stream>>>(s2, W3q, b3, nullptr, feat, B);
    k_head<<<B / 128, 256, 0, stream>>>(feat, Wmh, bm, noise, (float*)d_out, B);
}
